// TTTBase_80479097193098
// MI455X (gfx1250) — compile-verified, matrix operand order checked by reading, not run on hardware
//
#include <hip/hip_runtime.h>
#include <stdint.h>

// Problem constants (match reference)
#define B_   2
#define S_   2048
#define D_   2048
#define H_   32
#define HD_  64
#define MB_  16
#define N_   128
#define EPS_ 1e-5f

// ---------------- CDNA5 types ----------------
typedef __attribute__((ext_vector_type(16))) __bf16 v16bf;
typedef __attribute__((ext_vector_type(8)))  float  v8f;
typedef __attribute__((ext_vector_type(2)))  float  v2f;
typedef __attribute__((ext_vector_type(4)))  unsigned int u32x4;
typedef __attribute__((ext_vector_type(8)))  int i32x8;
typedef __attribute__((ext_vector_type(4)))  int i32x4;

#if __has_builtin(__builtin_amdgcn_wmma_f32_16x16x4_f32)
#define HAVE_WMMA_F32 1
#else
#define HAVE_WMMA_F32 0
#endif

#if __has_builtin(__builtin_amdgcn_tensor_load_to_lds) && __has_builtin(__builtin_amdgcn_s_wait_tensorcnt)
#define HAVE_TDM 1
#else
#define HAVE_TDM 0
#endif

struct Frag32B { u32x4 lo, hi; };

// 16-bit A-matrix 16x32 layout (ISA 7.12.2): lane half selects K-block of 8;
// elems 0..7 -> K = k0..k0+7, elems 8..15 -> K = k0+16..k0+23 (k0 = kk + 8*half).
__device__ __forceinline__ v16bf load_bf16_frag(const unsigned short* __restrict__ row, int k0) {
  Frag32B f;
  f.lo = *(const u32x4*)(row + k0);
  f.hi = *(const u32x4*)(row + k0 + 16);
  return __builtin_bit_cast(v16bf, f);
}

__device__ __forceinline__ unsigned short f32_to_bf16_rn(float x) {
  unsigned int u = __float_as_uint(x);
  u += 0x7FFFu + ((u >> 16) & 1u);   // round-to-nearest-even
  return (unsigned short)(u >> 16);
}

// wave32 butterfly reduction (result valid in all lanes)
__device__ __forceinline__ float wave_sum32(float v) {
#pragma unroll
  for (int m = 16; m > 0; m >>= 1) v += __shfl_xor(v, m, 32);
  return v;
}

#if HAVE_TDM
// Issue a TDM 2D tile load Global->LDS: tile rows x cols fp32, row stride
// `strideElems`, per CDNA5 ISA D# groups (8.3/8.4). Tracked by TENSORcnt.
// This toolchain exposes the 6-arg builtin (g0, g1, g2[4], g3[4], extra[8], cpol).
__device__ __forceinline__ void tdm_load_2d(unsigned int ldsAddr, unsigned long long ga,
                                            unsigned int rows, unsigned int cols,
                                            unsigned int strideElems) {
  u32x4 g0;
  g0[0] = 1u;                                        // count=1 (valid), no gather
  g0[1] = ldsAddr;                                   // lds_addr [63:32]
  g0[2] = (unsigned int)(ga & 0xFFFFFFFFu);          // global_addr lo
  g0[3] = (unsigned int)((ga >> 32) & 0x01FFFFFFu)   // global_addr [56:32]
        | (2u << 30);                                // type=2 ("image")
  i32x8 g1;
  g1[0] = (int)(2u << 16);                           // data_size=2 -> 4 bytes
  g1[1] = (int)((cols & 0xFFFFu) << 16);             // tensor_dim0[15:0]
  g1[2] = (int)(((cols >> 16) & 0xFFFFu) | ((rows & 0xFFFFu) << 16)); // td0 hi | td1 lo
  g1[3] = (int)(((rows >> 16) & 0xFFFFu) | ((cols & 0xFFFFu) << 16)); // td1 hi | tile_dim0
  g1[4] = (int)(rows & 0xFFFFu);                     // tile_dim1 (tile_dim2=0)
  g1[5] = (int)strideElems;                          // tensor_dim0_stride[31:0]
  g1[6] = 0;                                         // stride hi | dim1_stride lo
  g1[7] = 0;
  i32x4 z4 = {0, 0, 0, 0};
  i32x8 z8 = {0, 0, 0, 0, 0, 0, 0, 0};
  __builtin_amdgcn_tensor_load_to_lds(g0, g1, z4, z4, z8, 0);
}
__device__ __forceinline__ unsigned int lds_addr_of(const void* p) {
  return (unsigned int)(uintptr_t)p;   // flat LDS address: low 32 bits = LDS offset
}
#endif

// ---------------- fp32 -> bf16 convert ----------------
__global__ void cvt_f32_bf16(const float* __restrict__ in,
                             unsigned short* __restrict__ out, int n) {
  int i = blockIdx.x * blockDim.x + threadIdx.x;
  if (i < n) out[i] = f32_to_bf16_rn(in[i]);
}

// ---------------- WMMA GEMM: C[M,N] = A[M,K] * W[N,K]^T ----------------
__global__ __launch_bounds__(256)
void wmma_gemm_nt(const unsigned short* __restrict__ A,
                  const unsigned short* __restrict__ Wt,
                  float* __restrict__ C, int M, int Nn, int K) {
  const int lane = threadIdx.x & 31;
  const int wid  = threadIdx.x >> 5;
  const int wm   = wid >> 2;          // 0..1
  const int wn   = wid & 3;           // 0..3
  const int mBase = blockIdx.y * 128 + wm * 64;
  const int nBase = blockIdx.x * 128 + wn * 32;
  const int half  = lane >> 4;
  const int r16   = lane & 15;

  const unsigned short* aRow[4];
  const unsigned short* bRow[2];
#pragma unroll
  for (int i = 0; i < 4; i++) aRow[i] = A  + (size_t)(mBase + i * 16 + r16) * K;
#pragma unroll
  for (int j = 0; j < 2; j++) bRow[j] = Wt + (size_t)(nBase + j * 16 + r16) * K;

  v8f acc[4][2] = {};

  for (int kk = 0; kk < K; kk += 32) {
    const int k0 = kk + 8 * half;
    __builtin_prefetch(aRow[0] + k0 + 128, 0, 1);   // -> global_prefetch_b8
    __builtin_prefetch(bRow[0] + k0 + 128, 0, 1);
    v16bf a[4], b[2];
#pragma unroll
    for (int i = 0; i < 4; i++) a[i] = load_bf16_frag(aRow[i], k0);
#pragma unroll
    for (int j = 0; j < 2; j++) b[j] = load_bf16_frag(bRow[j], k0);
#pragma unroll
    for (int i = 0; i < 4; i++)
#pragma unroll
      for (int j = 0; j < 2; j++)
        acc[i][j] = __builtin_amdgcn_wmma_f32_16x16x32_bf16(
            false, a[i], false, b[j], (short)0, acc[i][j], false, false);
  }

  const int cn = r16, ch = half;
#pragma unroll
  for (int i = 0; i < 4; i++)
#pragma unroll
    for (int j = 0; j < 2; j++)
#pragma unroll
      for (int e = 0; e < 8; e++)
        C[(size_t)(mBase + i * 16 + e + 8 * ch) * Nn + (nBase + j * 16 + cn)] =
            acc[i][j][e];
}

// ---------------- RoPE + (B,S,H,HD) -> (B,H,S,HD) reshape ----------------
__global__ void rope_reshape(const float* __restrict__ in,     // (B,S,D)
                             const float* __restrict__ freqs,  // (S, HD/2)
                             float* __restrict__ out) {        // (B,H,S,HD)
  int idx = blockIdx.x * 256 + threadIdx.x;
  int p = idx & 31;
  int h = (idx >> 5) & 31;
  int s = (idx >> 10) & (S_ - 1);
  int b = idx >> 21;
  float fr = freqs[s * (HD_ / 2) + p];
  float cs = cosf(fr), sn = sinf(fr);
  const float* src = in + ((size_t)b * S_ + s) * D_ + h * HD_ + 2 * p;
  float x1 = src[0], x2 = src[1];
  float* dst = out + (((size_t)(b * H_ + h)) * S_ + s) * HD_ + 2 * p;
  dst[0] = x1 * cs - x2 * sn;
  dst[1] = x1 * sn + x2 * cs;
}

// ---------------- per-token learning rate ----------------
__global__ __launch_bounds__(256)
void lr_sigmoid(const float* __restrict__ x, const float* __restrict__ ilrW,
                const float* __restrict__ ilrb, float* __restrict__ lr) {
  int gw = blockIdx.x * 8 + (threadIdx.x >> 5);   // wave id over B*H*S
  int lane = threadIdx.x & 31;
  int s = gw & (S_ - 1);
  int h = (gw >> 11) & (H_ - 1);
  int b = gw >> 16;
  const float* xr = x + ((size_t)b * S_ + s) * D_;
  const float* w  = ilrW + (size_t)h * D_;
  float acc = 0.f;
  for (int k = lane; k < D_; k += 32) acc += xr[k] * w[k];
  acc = wave_sum32(acc);
  if (lane == 0) {
    float z = acc + ilrb[h];
    lr[gw] = (1.f / (1.f + expf(-z))) * (1.0f / (float)HD_);
  }
}

// ---------------- TTT scan: one block per (b,h), state in LDS ----------------
// f32 WMMA for XK@W, XQ@W and the rank-16 W update; TDM double-buffered X tiles.
__global__ __launch_bounds__(256)
void ttt_scan(const float* __restrict__ qr, const float* __restrict__ kr,
              const float* __restrict__ vr, const float* __restrict__ lrw,
              const float* __restrict__ lgs, const float* __restrict__ tg,
              const float* __restrict__ tb, const float* __restrict__ W0,
              const float* __restrict__ b0, float* __restrict__ outp) {
  __shared__ float Wm[64][65];
  __shared__ float bvec[64], gam[64], bet[64];
  __shared__ float Xq[2][16][64], Xk[2][16][64], Xv[2][16][64];
  __shared__ float Zm[16][64], Gr[16][64], Zb[16][64];
  __shared__ float At[16][16];
  __shared__ float lrv[16], gsv[16];

  const int b = blockIdx.x >> 5;
  const int h = blockIdx.x & 31;
  const int tid = threadIdx.x;
  const int lane = tid & 31, wid = tid >> 5;

  for (int t = tid; t < 4096; t += 256) Wm[t >> 6][t & 63] = W0[(size_t)h * 4096 + t];
  if (tid < 64) {
    bvec[tid] = b0[h * 64 + tid];
    gam[tid]  = tg[h * 64 + tid];
    bet[tid]  = tb[h * 64 + tid];
  }
  if (tid < 16) gsv[tid] = fmaxf(1.0f / (float)(tid + 1) + lgs[tid], 0.0f);

  const size_t headBase = ((size_t)(b * H_ + h)) * S_;

#if HAVE_TDM
  if (wid == 0) {   // prologue: DMA minibatch 0 into buffer 0
    tdm_load_2d(lds_addr_of(&Xq[0][0][0]), (unsigned long long)(uintptr_t)(qr + headBase * HD_), MB_, HD_, HD_);
    tdm_load_2d(lds_addr_of(&Xk[0][0][0]), (unsigned long long)(uintptr_t)(kr + headBase * HD_), MB_, HD_, HD_);
    tdm_load_2d(lds_addr_of(&Xv[0][0][0]), (unsigned long long)(uintptr_t)(vr + headBase * HD_), MB_, HD_, HD_);
  }
#endif
  __syncthreads();

  for (int n = 0; n < N_; n++) {
    const int s0 = n * MB_;
    const int cur = n & 1;
#if HAVE_TDM
    if (wid == 0) {
      __builtin_amdgcn_s_wait_tensorcnt((short)0);   // buffer `cur` resident
      if (n + 1 < N_) {                              // prefetch n+1 into other buffer
        const size_t nb = (headBase + (size_t)(n + 1) * MB_) * HD_;
        tdm_load_2d(lds_addr_of(&Xq[cur ^ 1][0][0]), (unsigned long long)(uintptr_t)(qr + nb), MB_, HD_, HD_);
        tdm_load_2d(lds_addr_of(&Xk[cur ^ 1][0][0]), (unsigned long long)(uintptr_t)(kr + nb), MB_, HD_, HD_);
        tdm_load_2d(lds_addr_of(&Xv[cur ^ 1][0][0]), (unsigned long long)(uintptr_t)(vr + nb), MB_, HD_, HD_);
      }
    }
#else
    {
      const float* qb = qr + (headBase + s0) * HD_;
      const float* kb = kr + (headBase + s0) * HD_;
      const float* vb = vr + (headBase + s0) * HD_;
      for (int t = tid; t < 1024; t += 256) {
        Xq[cur][t >> 6][t & 63] = qb[t];
        Xk[cur][t >> 6][t & 63] = kb[t];
        Xv[cur][t >> 6][t & 63] = vb[t];
      }
    }
#endif
    if (tid < 16) lrv[tid] = lrw[headBase + s0 + tid];
    __syncthreads();

    float (*XQ)[64] = Xq[cur];
    float (*XK)[64] = Xk[cur];
    float (*XV)[64] = Xv[cur];

    // ---- Z = XK@W + b (waves 0-3) and Zb = XQ@W + b (waves 4-7); Attn (all) ----
#if HAVE_WMMA_F32
    {
      const int tcol = (wid & 3) * 16;
      float (*Xsrc)[64] = (wid < 4) ? XK : XQ;
      float (*dst)[64]  = (wid < 4) ? Zm : Zb;
      const int m = lane & 15, hf = lane >> 4;
      v8f acc = {};
#pragma unroll
      for (int k0 = 0; k0 < 64; k0 += 4) {
        const int k = k0 + 2 * hf;
        v2f a, bb2;
        a[0] = Xsrc[m][k];        a[1] = Xsrc[m][k + 1];
        bb2[0] = Wm[k][tcol + m]; bb2[1] = Wm[k + 1][tcol + m];
        acc = __builtin_amdgcn_wmma_f32_16x16x4_f32(
            false, a, false, bb2, (short)0, acc, false, false);
      }
      const float bv = bvec[tcol + m];
#pragma unroll
      for (int e = 0; e < 8; e++) dst[e + 8 * hf][tcol + m] = acc[e] + bv;
    }
#else
    for (int t = tid; t < 1024; t += 256) {
      int i = t >> 6, c = t & 63;
      float a = bvec[c], a2 = bvec[c];
#pragma unroll 8
      for (int k = 0; k < 64; k++) { a += XK[i][k] * Wm[k][c]; a2 += XQ[i][k] * Wm[k][c]; }
      Zm[i][c] = a;
      Zb[i][c] = a2;
    }
#endif
    {
      int i = tid >> 4, j = tid & 15;
      float a = 0.f;
      if (j <= i) {
        for (int k = 0; k < 64; k++) a += XQ[i][k] * XK[j][k];
      }
      At[i][j] = a;
    }
    __syncthreads();

    // ---- grad = _ln_l2_bwd(Z, XV-XK, gamma, beta); each wave: 2 rows ----
    for (int r = wid * 2; r < wid * 2 + 2; r++) {
      float v0 = Zm[r][lane], v1 = Zm[r][lane + 32];
      float sm = wave_sum32(v0 + v1) * (1.f / 64.f);
      float sq = wave_sum32(v0 * v0 + v1 * v1) * (1.f / 64.f);
      float rs = rsqrtf(sq - sm * sm + EPS_);
      float zh0 = (v0 - sm) * rs, zh1 = (v1 - sm) * rs;
      float dz0 = (gam[lane] * zh0 + bet[lane] - (XV[r][lane] - XK[r][lane])) * gam[lane];
      float dz1 = (gam[lane + 32] * zh1 + bet[lane + 32] -
                   (XV[r][lane + 32] - XK[r][lane + 32])) * gam[lane + 32];
      float m1 = wave_sum32(dz0 + dz1) * (1.f / 64.f);
      float m2 = wave_sum32(dz0 * zh0 + dz1 * zh1) * (1.f / 64.f);
      Gr[r][lane]      = (dz0 - m1 - zh0 * m2) * rs;
      Gr[r][lane + 32] = (dz1 - m1 - zh1 * m2) * rs;
    }
    __syncthreads();

    // ---- Z_bar corrections: Zb -= [E*(Attn+1)]_tril @ grad ----
    for (int t = tid; t < 1024; t += 256) {
      int i = t >> 6, c = t & 63;
      float a = Zb[i][c];
      float gi = gsv[i];
      for (int j = 0; j <= i; j++) a -= gi * lrv[j] * (At[i][j] + 1.0f) * Gr[j][c];
      Zb[i][c] = a;
    }
    __syncthreads();

    // ---- y = XQ + _ln_fwd(Z_bar) ----
    for (int r = wid * 2; r < wid * 2 + 2; r++) {
      float v0 = Zb[r][lane], v1 = Zb[r][lane + 32];
      float sm = wave_sum32(v0 + v1) * (1.f / 64.f);
      float sq = wave_sum32(v0 * v0 + v1 * v1) * (1.f / 64.f);
      float rs = rsqrtf(sq - sm * sm + EPS_);
      float* o = outp + ((size_t)b * S_ + s0 + r) * D_ + h * HD_;
      o[lane]      = XQ[r][lane]      + gam[lane]      * ((v0 - sm) * rs) + bet[lane];
      o[lane + 32] = XQ[r][lane + 32] + gam[lane + 32] * ((v1 - sm) * rs) + bet[lane + 32];
    }

    // ---- W -= (last_eta*XK)^T @ grad ; b -= sum_j last_eta[j]*grad[j] ----
    const float le = gsv[MB_ - 1];
#if HAVE_WMMA_F32
    {
      const int m = lane & 15, hf = lane >> 4;
#pragma unroll
      for (int s = 0; s < 2; s++) {
        const int tt = wid * 2 + s;
        const int rB = (tt >> 2) * 16, cB = (tt & 3) * 16;
        v8f acc;
#pragma unroll
        for (int e = 0; e < 8; e++) acc[e] = Wm[rB + e + 8 * hf][cB + m];
#pragma unroll
        for (int k0 = 0; k0 < 16; k0 += 4) {
          const int k = k0 + 2 * hf;
          v2f a, bb2;
          a[0] = -le * lrv[k]     * XK[k][rB + m];   // fold -eta into A (f32 WMMA: CNeg only)
          a[1] = -le * lrv[k + 1] * XK[k + 1][rB + m];
          bb2[0] = Gr[k][cB + m];
          bb2[1] = Gr[k + 1][cB + m];
          acc = __builtin_amdgcn_wmma_f32_16x16x4_f32(
              false, a, false, bb2, (short)0, acc, false, false);
        }
#pragma unroll
        for (int e = 0; e < 8; e++) Wm[rB + e + 8 * hf][cB + m] = acc[e];
      }
    }
#else
    for (int t = tid; t < 4096; t += 256) {
      int r = t >> 6, c = t & 63;
      float a = 0.f;
#pragma unroll
      for (int j = 0; j < 16; j++) a += lrv[j] * XK[j][r] * Gr[j][c];
      Wm[r][c] -= le * a;
    }
#endif
    if (tid < 64) {
      float a = 0.f;
#pragma unroll
      for (int j = 0; j < 16; j++) a += lrv[j] * Gr[j][tid];
      bvec[tid] -= le * a;
    }
    __syncthreads();
  }
}

// ---------------- post layernorm -> bf16 ----------------
__global__ __launch_bounds__(256)
void post_ln_bf16(const float* __restrict__ in, const float* __restrict__ g,
                  const float* __restrict__ be, unsigned short* __restrict__ out) {
  __shared__ float red1[8], red2[8];
  const int row = blockIdx.x;
  const float* r = in + (size_t)row * D_;
  float s1 = 0.f, s2 = 0.f;
  for (int c = threadIdx.x; c < D_; c += 256) {
    float v = r[c];
    s1 += v; s2 += v * v;
  }
  s1 = wave_sum32(s1); s2 = wave_sum32(s2);
  int lane = threadIdx.x & 31, wid = threadIdx.x >> 5;
  if (lane == 0) { red1[wid] = s1; red2[wid] = s2; }
  __syncthreads();
  s1 = 0.f; s2 = 0.f;
#pragma unroll
  for (int i = 0; i < 8; i++) { s1 += red1[i]; s2 += red2[i]; }
  float mu = s1 * (1.f / (float)D_);
  float rs = rsqrtf(s2 * (1.f / (float)D_) - mu * mu + EPS_);
  unsigned short* o = out + (size_t)row * D_;
  for (int c = threadIdx.x; c < D_; c += 256)
    o[c] = f32_to_bf16_rn(g[c] * ((r[c] - mu) * rs) + be[c]);
}

// ---------------- launcher ----------------
extern "C" void kernel_launch(void* const* d_in, const int* in_sizes, int n_in,
                              void* d_out, int out_size, void* d_ws, size_t ws_size,
                              hipStream_t stream) {
  (void)in_sizes; (void)n_in; (void)out_size; (void)ws_size;
  const float* x    = (const float*)d_in[0];
  const float* posf = (const float*)d_in[1];
  const float* Wq   = (const float*)d_in[2];
  const float* Wk   = (const float*)d_in[3];
  const float* Wv   = (const float*)d_in[4];
  const float* Wo   = (const float*)d_in[5];
  const float* pg   = (const float*)d_in[6];
  const float* pb   = (const float*)d_in[7];
  const float* ilrW = (const float*)d_in[8];
  const float* ilrb = (const float*)d_in[9];
  const float* lgs  = (const float*)d_in[10];
  const float* tg   = (const float*)d_in[11];
  const float* tb   = (const float*)d_in[12];
  const float* W0   = (const float*)d_in[13];
  const float* b0   = (const float*)d_in[14];
  float* out = (float*)d_out;

  char* p = (char*)d_ws;
  auto carve = [&](size_t bytes) {
    char* r = p;
    p += (bytes + 255) & ~(size_t)255;
    return r;
  };
  const size_t BS = (size_t)B_ * S_;  // 4096
  unsigned short* xbf  = (unsigned short*)carve(BS * D_ * 2);
  unsigned short* wqbf = (unsigned short*)carve((size_t)D_ * D_ * 2);
  unsigned short* wkbf = (unsigned short*)carve((size_t)D_ * D_ * 2);
  unsigned short* wvbf = (unsigned short*)carve((size_t)D_ * D_ * 2);
  unsigned short* wobf = (unsigned short*)carve((size_t)D_ * D_ * 2);
  float* tmp = (float*)carve(BS * D_ * 4);   // GEMM output; later reused as out_pre
  float* q_r = (float*)carve(BS * D_ * 4);
  float* k_r = (float*)carve(BS * D_ * 4);
  float* v_r = (float*)carve(BS * D_ * 4);
  float* lrw = (float*)carve((size_t)B_ * H_ * S_ * 4);
  unsigned short* ybf = xbf;   // x_bf dead after the V projection GEMM
  float* outp = tmp;           // tmp dead after rope(v)

  // 1) bf16 conversions
  cvt_f32_bf16<<<(int)((BS * D_ + 255) / 256), 256, 0, stream>>>(x, xbf, (int)(BS * D_));
  cvt_f32_bf16<<<(D_ * D_ + 255) / 256, 256, 0, stream>>>(Wq, wqbf, D_ * D_);
  cvt_f32_bf16<<<(D_ * D_ + 255) / 256, 256, 0, stream>>>(Wk, wkbf, D_ * D_);
  cvt_f32_bf16<<<(D_ * D_ + 255) / 256, 256, 0, stream>>>(Wv, wvbf, D_ * D_);
  cvt_f32_bf16<<<(D_ * D_ + 255) / 256, 256, 0, stream>>>(Wo, wobf, D_ * D_);

  // 2) Q/K/V projections (bf16 WMMA) + RoPE/reshape
  dim3 gemmGrid(D_ / 128, (int)(BS / 128));   // (16, 32)
  int ropeBlocks = (B_ * S_ * H_ * (HD_ / 2)) / 256;
  wmma_gemm_nt<<<gemmGrid, 256, 0, stream>>>(xbf, wqbf, tmp, (int)BS, D_, D_);
  rope_reshape<<<ropeBlocks, 256, 0, stream>>>(tmp, posf, q_r);
  wmma_gemm_nt<<<gemmGrid, 256, 0, stream>>>(xbf, wkbf, tmp, (int)BS, D_, D_);
  rope_reshape<<<ropeBlocks, 256, 0, stream>>>(tmp, posf, k_r);
  wmma_gemm_nt<<<gemmGrid, 256, 0, stream>>>(xbf, wvbf, tmp, (int)BS, D_, D_);
  rope_reshape<<<ropeBlocks, 256, 0, stream>>>(tmp, posf, v_r);

  // 3) learning rates
  lr_sigmoid<<<(B_ * H_ * S_) / 8, 256, 0, stream>>>(x, ilrW, ilrb, lrw);

  // 4) sequential TTT scan (f32 WMMA + TDM double buffering)
  ttt_scan<<<B_ * H_, 256, 0, stream>>>(q_r, k_r, v_r, lrw, lgs, tg, tb, W0, b0, outp);

  // 5) post-LN -> bf16, then 6) output projection (bf16 WMMA) into d_out
  post_ln_bf16<<<(int)BS, 256, 0, stream>>>(outp, pg, pb, ybf);
  wmma_gemm_nt<<<gemmGrid, 256, 0, stream>>>(ybf, wobf, out, (int)BS, D_, D_);
}
